// CapsuleNetwork_6459630814026
// MI455X (gfx1250) — compile-verified, matrix operand order checked by reading, not run on hardware
//
#include <hip/hip_runtime.h>

typedef __bf16 bf16_t;
typedef __attribute__((ext_vector_type(16))) __bf16 v16bf;
typedef __attribute__((ext_vector_type(8)))  float  v8f;
typedef __attribute__((address_space(3))) char lds_char_t;

#define BATCH 128
#define SEQ_T 64
#define EMB   300
#define EMB_P 320
#define HID   256
#define GATES 1024
#define DA_   128
#define RCAP  8
#define SC_   32
#define AT_   16
#define H_PITCH 528   // 256 bf16 + 16B pad, bank-rotating row pitch

union V16Q { v16bf v; uint4 q[2]; };

__device__ __forceinline__ float sigmoidf_(float x) { return 1.f / (1.f + __expf(-x)); }

// async global->LDS 16B copy (CDNA5 TDM-adjacent path, tracked by ASYNCcnt).
// INST_OFFSET is applied to BOTH the LDS and global addresses (ISA 15.18.3).
__device__ __forceinline__ void async_copy16(void* lds_dst, const void* gsrc)
{
    unsigned lds_off = (unsigned)(uintptr_t)(lds_char_t*)lds_dst;
    unsigned long long ga = (unsigned long long)(uintptr_t)gsrc;
    asm volatile("global_load_async_to_lds_b128 %0, %1, off"
                 :: "v"(lds_off), "v"(ga) : "memory");
}
__device__ __forceinline__ void async_copy16_off16(void* lds_dst, const void* gsrc)
{
    unsigned lds_off = (unsigned)(uintptr_t)(lds_char_t*)lds_dst;
    unsigned long long ga = (unsigned long long)(uintptr_t)gsrc;
    asm volatile("global_load_async_to_lds_b128 %0, %1, off offset:16"
                 :: "v"(lds_off), "v"(ga) : "memory");
}
__device__ __forceinline__ void wait_async0()
{
    asm volatile("s_wait_asynccnt 0" ::: "memory");
}

// ---------------------------------------------------------------------------
// Embedding gather -> bf16, [T][B][EMB_P] with zero padding past EMB
// ---------------------------------------------------------------------------
__global__ void k_embed(const int* __restrict__ tokens, const float* __restrict__ emb,
                        bf16_t* __restrict__ X)
{
    const long total = (long)SEQ_T * BATCH * EMB_P;
    for (long idx = (long)blockIdx.x * blockDim.x + threadIdx.x; idx < total;
         idx += (long)gridDim.x * blockDim.x) {
        int col = (int)(idx % EMB_P);
        long row = idx / EMB_P;
        int b = (int)(row & (BATCH - 1));
        int t = (int)(row >> 7);
        float v = 0.f;
        if (col < EMB) {
            int tok = tokens[b * SEQ_T + t];
            v = emb[(long)tok * EMB + col];
        }
        X[idx] = (bf16_t)v;
    }
}

// ---------------------------------------------------------------------------
// Pack a weight matrix into the WMMA B-operand lane layout (bf16).
// Tile (kt,nt) = 32x16; storage: tile-major, within tile lane-major 16 elems.
// Lane<16: N=lane,   K in {0..7, 16..23};  lane>=16: N=lane-16, K in {8..15, 24..31}.
// src_is_kn=0: W is [O=N][I=K] row-major (torch weight), optionally gate-permuted
//              (n' = 4*j+gate  ->  o = (n&3)*256 + (n>>2)).
// src_is_kn=1: W is already [K][N] row-major (caps_w per r).
// ---------------------------------------------------------------------------
__global__ void k_pack_b(const float* __restrict__ W, long sW, int Kreal, int Nreal, int ldW,
                         int src_is_kn, int gate_perm, int Kp, int Np,
                         bf16_t* __restrict__ outp, long sOut)
{
    const long batch = blockIdx.y;
    const long total = (long)Kp * Np;
    const int NT = Np >> 4;
    for (long idx = (long)blockIdx.x * blockDim.x + threadIdx.x; idx < total;
         idx += (long)gridDim.x * blockDim.x) {
        long tile = idx >> 9;
        int within = (int)(idx & 511);
        int lane = within >> 4, e = within & 15;
        int nt = (int)(tile % NT);
        int kt = (int)(tile / NT);
        int n = nt * 16 + (lane & 15);
        int k = kt * 32 + (e & 7) + ((e >> 3) << 4) + ((lane >> 4) << 3);
        float v = 0.f;
        if (k < Kreal && n < Nreal) {
            if (src_is_kn) v = W[batch * sW + (long)k * ldW + n];
            else {
                int o = gate_perm ? ((n & 3) * 256 + (n >> 2)) : n;
                v = W[batch * sW + (long)o * ldW + k];
            }
        }
        outp[batch * sOut + idx] = (bf16_t)v;
    }
}

// ---------------------------------------------------------------------------
// Generic WMMA GEMM: C[M,N] = A[M,K](bf16,row-major) @ Bpacked[K,N] (+bias,act)
// Block = 256 threads (8 waves), tile 128x128, K step 32.
// A staged in LDS via GLOBAL_LOAD_ASYNC_TO_LDS_B128 (ASYNCcnt path).
// bias_mode: 0 none, 1 bias[n], 2 gate-permuted bias[(n&3)*256+(n>>2)]
// act: 0 none, 1 tanh
// ---------------------------------------------------------------------------
__global__ __launch_bounds__(256)
void k_gemm128(const bf16_t* __restrict__ A, long lda, long sA,
               const bf16_t* __restrict__ Bp, long sB, int NT,
               const float* __restrict__ bias, int bias_mode,
               float* __restrict__ C, long ldc, long sC,
               int K, int act)
{
    __shared__ __align__(16) bf16_t a_lds[128 * 40];   // pitch 40 elems = 80B (bank-rotating)
    const int tid  = threadIdx.x;
    const int lane = tid & 31;
    const int wave = tid >> 5;
    const int l15  = lane & 15;
    const int lh   = lane >> 4;
    const int mb   = wave >> 2;            // 0..1 -> 64-row half
    const int nb   = wave & 3;             // 0..3 -> 32-col slice
    const long batch = blockIdx.z;
    const long aBase = batch * sA + (long)(blockIdx.x * 128) * lda;
    const int  ntBase = blockIdx.y * 8 + nb * 2;
    const int  KT = K >> 5;

    const v8f vzero = {0.f,0.f,0.f,0.f,0.f,0.f,0.f,0.f};
    v8f acc[4][2];
#pragma unroll
    for (int mi = 0; mi < 4; ++mi)
#pragma unroll
        for (int ni = 0; ni < 2; ++ni) acc[mi][ni] = vzero;

    for (int kt = 0; kt < KT; ++kt) {
        __syncthreads();
        {   // stage A tile 128x32: each thread copies 32B with two async b128 ops
            const int row = tid >> 1, half = tid & 1;
            const bf16_t* ag = A + aBase + (long)row * lda + kt * 32 + half * 16;
            bf16_t* dst = a_lds + row * 40 + half * 16;
            async_copy16(dst, ag);
            async_copy16_off16(dst, ag);   // +16B on both LDS and global sides
            wait_async0();
        }
        __syncthreads();

        const bf16_t* bt = Bp + batch * sB + ((long)(kt * NT + ntBase) * 32 + lane) * 16;
        v16bf b0 = *(const v16bf*)bt;
        v16bf b1 = *(const v16bf*)(bt + 512);
        if (kt + 1 < KT)
            __builtin_prefetch(Bp + batch * sB + ((long)((kt + 1) * NT + ntBase) * 32 + lane) * 16, 0, 1);

#pragma unroll
        for (int mi = 0; mi < 4; ++mi) {
            const bf16_t* ap = a_lds + (mb * 64 + mi * 16 + l15) * 40 + lh * 8;
            V16Q u;
            u.q[0] = *(const uint4*)(ap);
            u.q[1] = *(const uint4*)(ap + 16);
            acc[mi][0] = __builtin_amdgcn_wmma_f32_16x16x32_bf16(false, u.v, false, b0,
                                                                (short)0, acc[mi][0], false, false);
            acc[mi][1] = __builtin_amdgcn_wmma_f32_16x16x32_bf16(false, u.v, false, b1,
                                                                (short)0, acc[mi][1], false, false);
        }
    }

#pragma unroll
    for (int mi = 0; mi < 4; ++mi) {
#pragma unroll
        for (int ni = 0; ni < 2; ++ni) {
            const int n = (ntBase + ni) * 16 + l15;
            float bv = 0.f;
            if (bias_mode == 1) bv = bias[n];
            else if (bias_mode == 2) bv = bias[(n & 3) * 256 + (n >> 2)];
#pragma unroll
            for (int r = 0; r < 8; ++r) {
                const long row = (long)blockIdx.x * 128 + mb * 64 + mi * 16 + lh * 8 + r;
                float v = acc[mi][ni][r] + bv;
                if (act == 1) v = tanhf(v);
                C[batch * sC + row * ldc + n] = v;
            }
        }
    }
}

// ---------------------------------------------------------------------------
// Recurrent bidirectional LSTM layer (input projections precomputed in Z).
// Gate columns are permuted n' = 4*j + gate so each wave owns complete gate
// quads; c-state stays in registers, h double-buffered in LDS (bf16).
// grid = 2 blocks (dir 0 = forward, 1 = backward), block = 1024 threads.
// ---------------------------------------------------------------------------
__global__ __launch_bounds__(1024)
void k_lstm(const float* __restrict__ Zf, const float* __restrict__ Zb,
            const bf16_t* __restrict__ Whh_f, const bf16_t* __restrict__ Whh_b,
            bf16_t* __restrict__ xout)
{
    extern __shared__ char smem[];
    const int dir = blockIdx.x;
    const float*  Z   = dir ? Zb : Zf;
    const bf16_t* Whh = dir ? Whh_b : Whh_f;
    const int col_off = dir * HID;

    const int tid  = threadIdx.x;
    const int lane = tid & 31;
    const int wave = tid >> 5;                // 0..31, owns 32 gate columns
    const int l15  = lane & 15;
    const int lh   = lane >> 4;
    const int grpb = lane & ~3;
    const int gate = lane & 3;                // i,f,g,o
    const int nt0  = wave * 2;

    char* hbuf0 = smem;
    char* hbuf1 = smem + BATCH * H_PITCH;

    for (int i = tid * 4; i < BATCH * H_PITCH; i += 1024 * 4)
        *(unsigned int*)(hbuf0 + i) = 0u;     // h(-1) = 0

    float c_reg[32];
#pragma unroll
    for (int i = 0; i < 32; ++i) c_reg[i] = 0.f;

    for (int t = 0; t < SEQ_T; ++t) {
        __syncthreads();
        const int tmem = dir ? (SEQ_T - 1 - t) : t;
        const float* Zt = Z + (long)tmem * BATCH * GATES;
        const char*  hin  = (t & 1) ? hbuf1 : hbuf0;
        char*        hout = (t & 1) ? hbuf0 : hbuf1;

#pragma unroll
        for (int mt = 0; mt < 8; ++mt) {
            v8f acc0, acc1;
            {   // seed accumulators with precomputed input projection (+bias)
                const int n0 = nt0 * 16 + l15;
                const int rowbase = mt * 16 + lh * 8;
#pragma unroll
                for (int r = 0; r < 8; ++r) {
                    const float* zr = Zt + (long)(rowbase + r) * GATES + n0;
                    acc0[r] = zr[0];
                    acc1[r] = zr[16];
                }
            }
#pragma unroll
            for (int kt = 0; kt < 8; ++kt) {  // K = 256
                const char* rp = hin + (mt * 16 + l15) * H_PITCH + kt * 64 + lh * 16;
                V16Q u;
                u.q[0] = *(const uint4*)rp;
                u.q[1] = *(const uint4*)(rp + 32);
                const bf16_t* bp = Whh + ((long)(kt * 64 + nt0) * 32 + lane) * 16;
                v16bf b0 = *(const v16bf*)bp;
                v16bf b1 = *(const v16bf*)(bp + 512);
                acc0 = __builtin_amdgcn_wmma_f32_16x16x32_bf16(false, u.v, false, b0,
                                                               (short)0, acc0, false, false);
                acc1 = __builtin_amdgcn_wmma_f32_16x16x32_bf16(false, u.v, false, b1,
                                                               (short)0, acc1, false, false);
            }
            // gate nonlinearities + cell/hidden update
#pragma unroll
            for (int q = 0; q < 2; ++q) {
                v8f z = q ? acc1 : acc0;
                float actv[8];
#pragma unroll
                for (int r = 0; r < 8; ++r) {
                    float x = z[r];
                    actv[r] = (gate == 2) ? tanhf(x) : sigmoidf_(x);
                }
                const int j = (nt0 + q) * 4 + (l15 >> 2);
#pragma unroll
                for (int r = 0; r < 8; ++r) {
                    float vi = __shfl(actv[r], grpb + 0, 32);
                    float vf = __shfl(actv[r], grpb + 1, 32);
                    float vg = __shfl(actv[r], grpb + 2, 32);
                    float vo = __shfl(actv[r], grpb + 3, 32);
                    if ((r >> 1) == gate) {   // lane keeps c for 2 of the 8 rows
                        const int slot = (mt * 2 + q) * 2 + (r & 1);
                        float cn = vf * c_reg[slot] + vi * vg;
                        c_reg[slot] = cn;
                        float h = vo * tanhf(cn);
                        bf16_t hb = (bf16_t)h;
                        const int brow = mt * 16 + lh * 8 + r;
                        *(bf16_t*)(hout + brow * H_PITCH + j * 2) = hb;
                        xout[((long)tmem * BATCH + brow) * 512 + col_off + j] = hb;
                    }
                }
            }
        }
    }
}

// ---------------------------------------------------------------------------
// attention scores: att[b][r][t] = <hbar[t*B+b,:], ws2[r,:]>   (tiny FLOPs)
// ---------------------------------------------------------------------------
__global__ void k_att(const float* __restrict__ hbar, const float* __restrict__ ws2,
                      float* __restrict__ att)
{
    const int total = BATCH * RCAP * SEQ_T;
    for (int idx = blockIdx.x * blockDim.x + threadIdx.x; idx < total;
         idx += gridDim.x * blockDim.x) {
        int t = idx % SEQ_T;
        int r = (idx / SEQ_T) % RCAP;
        int b = idx / (SEQ_T * RCAP);
        const float* hm = hbar + ((long)t * BATCH + b) * DA_;
        const float* wr = ws2 + r * DA_;
        float s = 0.f;
        for (int d = 0; d < DA_; ++d) s += hm[d] * wr[d];
        att[idx] = s;
    }
}

// sent[b][r][u] = sum_t att[b][r][t] * outp[t*B+b][u]
__global__ void k_sent(const float* __restrict__ att, const bf16_t* __restrict__ x2,
                       float* __restrict__ sent)
{
    const long total = (long)BATCH * RCAP * 512;
    for (long idx = (long)blockIdx.x * blockDim.x + threadIdx.x; idx < total;
         idx += (long)gridDim.x * blockDim.x) {
        int u = (int)(idx & 511);
        int r = (int)((idx >> 9) & 7);
        int b = (int)(idx >> 12);
        const float* ab = att + ((long)b * RCAP + r) * SEQ_T;
        float s = 0.f;
        for (int t = 0; t < SEQ_T; ++t)
            s += ab[t] * (float)x2[((long)t * BATCH + b) * 512 + u];
        sent[idx] = s;
    }
}

__global__ void k_f2bf(const float* __restrict__ in, bf16_t* __restrict__ outp, long n)
{
    for (long i = (long)blockIdx.x * blockDim.x + threadIdx.x; i < n;
         i += (long)gridDim.x * blockDim.x)
        outp[i] = (bf16_t)in[i];
}

// ---------------------------------------------------------------------------
// Dynamic routing (3 iters) + final capsule norms. One block per batch elem.
// ---------------------------------------------------------------------------
__global__ __launch_bounds__(512)
void k_routing(const float* __restrict__ votes, float* __restrict__ outp)
{
    __shared__ float vs[RCAP * SC_ * AT_];
    __shared__ float logits[RCAP * SC_];
    __shared__ float route[RCAP * SC_];
    __shared__ float pre[SC_ * AT_];
    __shared__ float scl[SC_];
    const int b = blockIdx.x;
    const int tid = threadIdx.x;
    for (int i = tid; i < RCAP * SC_ * AT_; i += 512)
        vs[i] = votes[(long)b * (RCAP * SC_ * AT_) + i];
    if (tid < RCAP * SC_) logits[tid] = 0.f;
    __syncthreads();
    for (int it = 0; it < 3; ++it) {
        if (tid < RCAP) {
            const int r = tid;
            float mx = -3.0e38f;
            for (int c = 0; c < SC_; ++c) mx = fmaxf(mx, logits[r * SC_ + c]);
            float s = 0.f;
            for (int c = 0; c < SC_; ++c) {
                float e = __expf(logits[r * SC_ + c] - mx);
                route[r * SC_ + c] = e; s += e;
            }
            float inv = 1.f / s;
            for (int c = 0; c < SC_; ++c) route[r * SC_ + c] *= inv;
        }
        __syncthreads();
        {
            const int c = tid >> 4, a = tid & 15;
            float p = 0.f;
            for (int r = 0; r < RCAP; ++r) p += route[r * SC_ + c] * vs[(r * SC_ + c) * AT_ + a];
            pre[c * AT_ + a] = p;
        }
        __syncthreads();
        if (tid < SC_) {
            float n2 = 0.f;
            for (int a = 0; a < AT_; ++a) { float x = pre[tid * AT_ + a]; n2 += x * x; }
            scl[tid] = sqrtf(n2) / (0.5f + n2);   // squash: pre/nrm * n2/(0.5+n2)
        }
        __syncthreads();
        pre[tid] *= scl[tid >> 4];
        __syncthreads();
        if (tid < RCAP * SC_) {
            const int r = tid / SC_, c = tid % SC_;
            float d = 0.f;
            for (int a = 0; a < AT_; ++a) d += vs[(r * SC_ + c) * AT_ + a] * pre[c * AT_ + a];
            logits[tid] += d;
        }
        __syncthreads();
    }
    if (tid < SC_) {
        float n2 = 0.f;
        for (int a = 0; a < AT_; ++a) { float x = pre[tid * AT_ + a]; n2 += x * x; }
        outp[b * SC_ + tid] = sqrtf(n2);
    }
}

// ---------------------------------------------------------------------------
extern "C" void kernel_launch(void* const* d_in, const int* in_sizes, int n_in,
                              void* d_out, int out_size, void* d_ws, size_t ws_size,
                              hipStream_t stream)
{
    (void)in_sizes; (void)n_in; (void)out_size; (void)ws_size;
    const int*   tokens  = (const int*)  d_in[0];
    // d_in[1] = lengths, all == T (packing is a no-op)
    const float* emb     = (const float*)d_in[2];
    const float* w_ih_f0 = (const float*)d_in[3];
    const float* w_hh_f0 = (const float*)d_in[4];
    const float* b_f0    = (const float*)d_in[5];
    const float* w_ih_b0 = (const float*)d_in[6];
    const float* w_hh_b0 = (const float*)d_in[7];
    const float* b_b0    = (const float*)d_in[8];
    const float* w_ih_f1 = (const float*)d_in[9];
    const float* w_hh_f1 = (const float*)d_in[10];
    const float* b_f1    = (const float*)d_in[11];
    const float* w_ih_b1 = (const float*)d_in[12];
    const float* w_hh_b1 = (const float*)d_in[13];
    const float* b_b1    = (const float*)d_in[14];
    const float* ws1     = (const float*)d_in[15];
    const float* ws2     = (const float*)d_in[16];
    const float* caps_w  = (const float*)d_in[17];
    float* out = (float*)d_out;

    size_t off = 0;
    auto alloc = [&](size_t bytes) -> void* {
        off = (off + 255) & ~(size_t)255;
        void* p = (char*)d_ws + off;
        off += bytes;
        return p;
    };

    const long MROWS = (long)SEQ_T * BATCH;          // 8192
    bf16_t* Xemb   = (bf16_t*)alloc(MROWS * EMB_P * 2);
    bf16_t* Pih_f0 = (bf16_t*)alloc((size_t)EMB_P * GATES * 2);
    bf16_t* Pih_b0 = (bf16_t*)alloc((size_t)EMB_P * GATES * 2);
    bf16_t* Phh_f0 = (bf16_t*)alloc((size_t)HID * GATES * 2);
    bf16_t* Phh_b0 = (bf16_t*)alloc((size_t)HID * GATES * 2);
    bf16_t* Pih_f1 = (bf16_t*)alloc((size_t)512 * GATES * 2);
    bf16_t* Pih_b1 = (bf16_t*)alloc((size_t)512 * GATES * 2);
    bf16_t* Phh_f1 = (bf16_t*)alloc((size_t)HID * GATES * 2);
    bf16_t* Phh_b1 = (bf16_t*)alloc((size_t)HID * GATES * 2);
    bf16_t* Pws1   = (bf16_t*)alloc((size_t)512 * DA_ * 2);
    bf16_t* Pcaps  = (bf16_t*)alloc((size_t)RCAP * 512 * 512 * 2);
    float*  Zf     = (float*) alloc(MROWS * GATES * 4);   // reused by layers 0 & 1
    float*  Zb     = (float*) alloc(MROWS * GATES * 4);
    bf16_t* x1     = (bf16_t*)alloc(MROWS * 512 * 2);
    bf16_t* x2     = (bf16_t*)alloc(MROWS * 512 * 2);
    float*  hbar   = (float*) alloc(MROWS * DA_ * 4);
    float*  att    = (float*) alloc((size_t)BATCH * RCAP * SEQ_T * 4);
    float*  sent   = (float*) alloc((size_t)BATCH * RCAP * 512 * 4);
    bf16_t* sentbf = (bf16_t*)alloc((size_t)BATCH * RCAP * 512 * 2);
    float*  votes  = (float*) alloc((size_t)BATCH * RCAP * 512 * 4);

    auto pack = [&](const float* W, long sW, int Kreal, int Nreal, int ldW, int kn, int gp,
                    int Kp, int Np, bf16_t* dst, long sOut, int batches) {
        k_pack_b<<<dim3(512, batches), 256, 0, stream>>>(W, sW, Kreal, Nreal, ldW, kn, gp,
                                                         Kp, Np, dst, sOut);
    };
    auto gemm = [&](const bf16_t* A, long lda, long sA, const bf16_t* Bp, long sB, int NT,
                    const float* bias, int bmode, float* Cp, long ldc, long sC,
                    int M, int N, int K, int act, int batches) {
        dim3 g(M / 128, N / 128, batches);
        k_gemm128<<<g, 256, 0, stream>>>(A, lda, sA, Bp, sB, NT, bias, bmode,
                                         Cp, ldc, sC, K, act);
    };

    // ---- prep: embedding + weight packing (bf16, WMMA B layout) ----
    k_embed<<<1024, 256, 0, stream>>>(tokens, emb, Xemb);
    pack(w_ih_f0, 0, EMB, GATES, EMB, 0, 1, EMB_P, GATES, Pih_f0, 0, 1);
    pack(w_ih_b0, 0, EMB, GATES, EMB, 0, 1, EMB_P, GATES, Pih_b0, 0, 1);
    pack(w_hh_f0, 0, HID, GATES, HID, 0, 1, HID,   GATES, Phh_f0, 0, 1);
    pack(w_hh_b0, 0, HID, GATES, HID, 0, 1, HID,   GATES, Phh_b0, 0, 1);
    pack(w_ih_f1, 0, 512, GATES, 512, 0, 1, 512,   GATES, Pih_f1, 0, 1);
    pack(w_ih_b1, 0, 512, GATES, 512, 0, 1, 512,   GATES, Pih_b1, 0, 1);
    pack(w_hh_f1, 0, HID, GATES, HID, 0, 1, HID,   GATES, Phh_f1, 0, 1);
    pack(w_hh_b1, 0, HID, GATES, HID, 0, 1, HID,   GATES, Phh_b1, 0, 1);
    pack(ws1,    0, 512, DA_, 512, 0, 0, 512, DA_, Pws1, 0, 1);
    pack(caps_w, (long)512 * 512, 512, 512, 512, 1, 0, 512, 512, Pcaps, (long)512 * 512, RCAP);

    // ---- layer 0: input projections (WMMA) + recurrence ----
    gemm(Xemb, EMB_P, 0, Pih_f0, 0, GATES / 16, b_f0, 2, Zf, GATES, 0, 8192, GATES, EMB_P, 0, 1);
    gemm(Xemb, EMB_P, 0, Pih_b0, 0, GATES / 16, b_b0, 2, Zb, GATES, 0, 8192, GATES, EMB_P, 0, 1);
    k_lstm<<<2, 1024, 2 * BATCH * H_PITCH, stream>>>(Zf, Zb, Phh_f0, Phh_b0, x1);

    // ---- layer 1 ----
    gemm(x1, 512, 0, Pih_f1, 0, GATES / 16, b_f1, 2, Zf, GATES, 0, 8192, GATES, 512, 0, 1);
    gemm(x1, 512, 0, Pih_b1, 0, GATES / 16, b_b1, 2, Zb, GATES, 0, 8192, GATES, 512, 0, 1);
    k_lstm<<<2, 1024, 2 * BATCH * H_PITCH, stream>>>(Zf, Zb, Phh_f1, Phh_b1, x2);

    // ---- attention pooling ----
    gemm(x2, 512, 0, Pws1, 0, DA_ / 16, nullptr, 0, hbar, DA_, 0, 8192, DA_, 512, 1, 1);
    k_att<<<256, 256, 0, stream>>>(hbar, ws2, att);
    k_sent<<<2048, 256, 0, stream>>>(att, x2, sent);
    k_f2bf<<<1024, 256, 0, stream>>>(sent, sentbf, (long)BATCH * RCAP * 512);

    // ---- capsule votes (batched over r) + dynamic routing ----
    gemm(sentbf, (long)RCAP * 512, 512, Pcaps, (long)512 * 512, 512 / 16,
         nullptr, 0, votes, (long)RCAP * 512, 512, 128, 512, 512, 0, RCAP);
    k_routing<<<128, 512, 0, stream>>>(votes, out);
}
